// SAttention_5987184411303
// MI455X (gfx1250) — 8-bit matrix operand order repaired and checked by reading, compiled, not run on hardware
//
#include <hip/hip_runtime.h>
#include <stdint.h>

#define B_   2
#define N_   2048
#define C_   768
#define H_   12
#define D_   64
#define M_   (B_*N_)
#define O3_  (3*C_)
#define SCALE_ 0.125f
#define C2_  0.18033688011112042f

typedef __attribute__((ext_vector_type(8)))  int    v8i;
typedef __attribute__((ext_vector_type(8)))  float  v8f;
typedef __attribute__((ext_vector_type(16))) __bf16 v16bf;
typedef int v4i_raw __attribute__((vector_size(16)));
typedef __attribute__((address_space(1))) v4i_raw* gptr_b128;
typedef __attribute__((address_space(3))) v4i_raw* lptr_b128;

#if defined(__has_builtin)
#if __has_builtin(__builtin_amdgcn_global_load_async_to_lds_b128) && \
    __has_builtin(__builtin_amdgcn_s_wait_asynccnt)
#define HAVE_ASYNC 1
#endif
#endif
#ifndef HAVE_ASYNC
#define HAVE_ASYNC 0
#endif

__device__ __forceinline__ void cp_tile_4k(const int8_t* __restrict__ g,
                                           int8_t* __restrict__ l, int lane) {
#if HAVE_ASYNC
  #pragma unroll
  for (int i = 0; i < 8; ++i)
    __builtin_amdgcn_global_load_async_to_lds_b128(
        (gptr_b128)(g + i * 512 + lane * 16),
        (lptr_b128)(l + i * 512 + lane * 16),
        0, 0);
#else
  #pragma unroll
  for (int i = 0; i < 8; ++i)
    *(int4*)(l + i * 512 + lane * 16) = *(const int4*)(g + i * 512 + lane * 16);
#endif
}
__device__ __forceinline__ void cp_wait() {
#if HAVE_ASYNC
  __builtin_amdgcn_s_wait_asynccnt(0);
#endif
}
__device__ __forceinline__ void lds_fence() {
  __asm volatile("s_wait_dscnt 0" ::: "memory");
}

__global__ __launch_bounds__(256)
void cvt_f32_bf16_kernel(const float* __restrict__ s, __bf16* __restrict__ d, int n) {
  int i = blockIdx.x * 256 + threadIdx.x;
  if (i < n) d[i] = (__bf16)s[i];
}

__device__ __forceinline__ v16bf load_bf16_frag(const __bf16* __restrict__ rowp,
                                                int k0, int hi) {
  union { int4 q[2]; v16bf v; } u;
  const char* p = (const char*)(rowp + k0) + hi * 16;
  u.q[0] = *(const int4*)(p);
  u.q[1] = *(const int4*)(p + 32);
  return u.v;
}

__device__ __forceinline__ void gemm32x32_bf16(const __bf16* __restrict__ A,
                                               const __bf16* __restrict__ Bm,
                                               int m0, int o0, int K,
                                               v8f acc[2][2]) {
  const int lane = threadIdx.x & 31;
  const int r = lane & 15, hi = lane >> 4;
  for (int k0 = 0; k0 < K; k0 += 32) {
    v16bf a0 = load_bf16_frag(A  + (size_t)(m0 + r)      * K, k0, hi);
    v16bf a1 = load_bf16_frag(A  + (size_t)(m0 + 16 + r) * K, k0, hi);
    v16bf b0 = load_bf16_frag(Bm + (size_t)(o0 + r)      * K, k0, hi);
    v16bf b1 = load_bf16_frag(Bm + (size_t)(o0 + 16 + r) * K, k0, hi);
    acc[0][0] = __builtin_amdgcn_wmma_f32_16x16x32_bf16(false, a0, false, b0, (short)0, acc[0][0], false, false);
    acc[0][1] = __builtin_amdgcn_wmma_f32_16x16x32_bf16(false, a0, false, b1, (short)0, acc[0][1], false, false);
    acc[1][0] = __builtin_amdgcn_wmma_f32_16x16x32_bf16(false, a1, false, b0, (short)0, acc[1][0], false, false);
    acc[1][1] = __builtin_amdgcn_wmma_f32_16x16x32_bf16(false, a1, false, b1, (short)0, acc[1][1], false, false);
  }
}

__global__ __launch_bounds__(32)
void qkv_wmma_kernel(const __bf16* __restrict__ xb, const __bf16* __restrict__ wb,
                     int8_t* __restrict__ qf, int8_t* __restrict__ kf,
                     int8_t* __restrict__ vf) {
  const int lane = threadIdx.x;
  const int r = lane & 15, hi = lane >> 4;
  const int m0 = blockIdx.x * 32;
  const int o0 = blockIdx.y * 32;
  v8f acc[2][2] = {};
  gemm32x32_bf16(xb, wb, m0, o0, C_, acc);

  #pragma unroll
  for (int i = 0; i < 2; ++i)
    #pragma unroll
    for (int j = 0; j < 2; ++j)
      #pragma unroll
      for (int jr = 0; jr < 8; ++jr) {
        float f = acc[i][j][jr];
        int t = (f >= 0.5f) ? 1 : ((f < -0.5f) ? -1 : 0);
        int m = m0 + i * 16 + jr + 8 * hi;
        int o = o0 + j * 16 + r;
        int three = o / C_;
        int rem = o - three * C_;
        int h = rem >> 6, d = rem & 63;
        int b = m >> 11, n = m & (N_ - 1);
        int bh = b * H_ + h;
        int rr = n & 15;
        if (three == 0) {
          int ln = rr + ((((unsigned)d >> 3) & 1) << 4);
          int dw = (((unsigned)d >> 4) << 1) | (((unsigned)d >> 2) & 1);
          qf[((size_t)bh * (N_/16) + (n >> 4)) * 1024 + ln * 32 + dw * 4 + (d & 3)] = (int8_t)t;
        } else if (three == 1) {
          int ln = rr + ((((unsigned)d >> 3) & 1) << 4);
          int dw = (((unsigned)d >> 4) << 1) | (((unsigned)d >> 2) & 1);
          kf[((size_t)bh * (N_/16) + (n >> 4)) * 1024 + ln * 32 + dw * 4 + (d & 3)] = (int8_t)t;
        } else {
          int rc = d & 15, nl = n & 63;
          int ln = rc + ((((unsigned)nl >> 3) & 1) << 4);
          int dw = (((unsigned)nl >> 4) << 1) | (((unsigned)nl >> 2) & 1);
          vf[(((size_t)bh * (N_/64) + (n >> 6)) * 4 + (d >> 4)) * 1024 + ln * 32 + dw * 4 + (nl & 3)] = (int8_t)t;
        }
      }
}

__global__ __launch_bounds__(128)
void attn_kernel(const int8_t* __restrict__ qf, const int8_t* __restrict__ kf,
                 const int8_t* __restrict__ vf, __bf16* __restrict__ y) {
  __shared__ __align__(32) int8_t kbuf[2][4096];
  __shared__ __align__(32) int8_t vbuf[2][4096];
  __shared__ unsigned pbuf[4][32 * 16];
  const int tid = threadIdx.x;
  const int lane = tid & 31, wid = tid >> 5;
  const int r = lane & 15, hi = lane >> 4;
  const int bh  = blockIdx.x >> 4;
  const int blk = blockIdx.x & 15;
  const int b = bh / H_, h = bh % H_;

  const int8_t* kbase = kf + (size_t)bh * (N_/16) * 1024;
  const int8_t* vbase = vf + (size_t)bh * (N_/64) * 4096;
  const v8i zero = {0, 0, 0, 0, 0, 0, 0, 0};

  v8i aq[2];
  #pragma unroll
  for (int t = 0; t < 2; ++t)
    aq[t] = *(const v8i*)(qf + ((size_t)bh * (N_/16) + blk * 8 + wid * 2 + t) * 1024 + lane * 32);

  float lsum[2][8];
  #pragma unroll
  for (int t = 0; t < 2; ++t)
    #pragma unroll
    for (int j = 0; j < 8; ++j) lsum[t][j] = 0.0f;

  if (wid == 0) { cp_tile_4k(kbase, kbuf[0], lane); cp_wait(); }
  __syncthreads();
  for (int ch = 0; ch < 32; ++ch) {
    const int cur = ch & 1, nxt = cur ^ 1;
    if (wid == 0 && ch + 1 < 32) cp_tile_4k(kbase + (size_t)(ch + 1) * 4096, kbuf[nxt], lane);
    #pragma unroll
    for (int sub = 0; sub < 4; ++sub) {
      v8i bk = *(const v8i*)(kbuf[cur] + sub * 1024 + lane * 32);
      #pragma unroll
      for (int t = 0; t < 2; ++t) {
        v8i s = __builtin_amdgcn_wmma_i32_16x16x64_iu8(true, aq[t], true, bk, zero, false, false);
        #pragma unroll
        for (int j = 0; j < 8; ++j)
          lsum[t][j] += __builtin_amdgcn_exp2f((float)s[j] * C2_);
      }
    }
    if (wid == 0) cp_wait();
    __syncthreads();
  }
  #pragma unroll
  for (int t = 0; t < 2; ++t)
    #pragma unroll
    for (int j = 0; j < 8; ++j) {
      float v = lsum[t][j];
      v += __shfl_xor(v, 1);
      v += __shfl_xor(v, 2);
      v += __shfl_xor(v, 4);
      v += __shfl_xor(v, 8);
      lsum[t][j] = v;
    }

  if (wid == 0) { cp_tile_4k(kbase, kbuf[0], lane); cp_tile_4k(vbase, vbuf[0], lane); cp_wait(); }
  __syncthreads();
  v8i oacc[2][4] = {{zero, zero, zero, zero}, {zero, zero, zero, zero}};
  for (int ch = 0; ch < 32; ++ch) {
    const int cur = ch & 1, nxt = cur ^ 1;
    if (wid == 0 && ch + 1 < 32) {
      cp_tile_4k(kbase + (size_t)(ch + 1) * 4096, kbuf[nxt], lane);
      cp_tile_4k(vbase + (size_t)(ch + 1) * 4096, vbuf[nxt], lane);
    }
    #pragma unroll
    for (int sub = 0; sub < 4; ++sub) {
      v8i bk = *(const v8i*)(kbuf[cur] + sub * 1024 + lane * 32);
      #pragma unroll
      for (int t = 0; t < 2; ++t) {
        v8i s = __builtin_amdgcn_wmma_i32_16x16x64_iu8(true, aq[t], true, bk, zero, false, false);
        #pragma unroll
        for (int j = 0; j < 8; ++j) {
          float e = __builtin_amdgcn_exp2f((float)s[j] * C2_);
          unsigned char bit = (e + e >= lsum[t][j]) ? 1u : 0u;
          ((unsigned char*)pbuf[wid])[(t * 16 + j + 8 * hi) * 64 + sub * 16 + r] = bit;
        }
      }
    }
    lds_fence();
    v8i pa[2];
    #pragma unroll
    for (int t = 0; t < 2; ++t) {
      unsigned base = (unsigned)(t * 16 + r) * 16 + (unsigned)hi * 2;
      pa[t][0] = (int)pbuf[wid][base +  0]; pa[t][1] = (int)pbuf[wid][base +  1];
      pa[t][2] = (int)pbuf[wid][base +  4]; pa[t][3] = (int)pbuf[wid][base +  5];
      pa[t][4] = (int)pbuf[wid][base +  8]; pa[t][5] = (int)pbuf[wid][base +  9];
      pa[t][6] = (int)pbuf[wid][base + 12]; pa[t][7] = (int)pbuf[wid][base + 13];
    }
    #pragma unroll
    for (int dt = 0; dt < 4; ++dt) {
      v8i bv = *(const v8i*)(vbuf[cur] + dt * 1024 + lane * 32);
      #pragma unroll
      for (int t = 0; t < 2; ++t)
        oacc[t][dt] = __builtin_amdgcn_wmma_i32_16x16x64_iu8(false, pa[t], true, bv, oacc[t][dt], false, false);
    }
    if (wid == 0) cp_wait();
    __syncthreads();
  }

  #pragma unroll
  for (int t = 0; t < 2; ++t)
    #pragma unroll
    for (int dt = 0; dt < 4; ++dt)
      #pragma unroll
      for (int j = 0; j < 8; ++j) {
        int val = oacc[t][dt][j];
        int tern = (val >= 1) - (val <= -1);
        int nq = blk * 128 + wid * 32 + t * 16 + j + 8 * hi;
        int cc = h * 64 + dt * 16 + r;
        y[(size_t)(b * N_ + nq) * C_ + cc] = (__bf16)(float)tern;
      }
}

__global__ __launch_bounds__(32)
void proj_wmma_kernel(const __bf16* __restrict__ yb, const __bf16* __restrict__ wb,
                      float* __restrict__ out) {
  const int lane = threadIdx.x;
  const int r = lane & 15, hi = lane >> 4;
  const int m0 = blockIdx.x * 32;
  const int o0 = blockIdx.y * 32;
  v8f acc[2][2] = {};
  gemm32x32_bf16(yb, wb, m0, o0, C_, acc);
  #pragma unroll
  for (int i = 0; i < 2; ++i)
    #pragma unroll
    for (int j = 0; j < 2; ++j)
      #pragma unroll
      for (int jr = 0; jr < 8; ++jr) {
        float f = acc[i][j][jr];
        float tern = (f >= 0.5f) ? 1.0f : ((f < -0.5f) ? -1.0f : 0.0f);
        int m = m0 + i * 16 + jr + 8 * hi;
        int o = o0 + j * 16 + r;
        out[(size_t)m * C_ + o] = tern;
      }
}

extern "C" void kernel_launch(void* const* d_in, const int* in_sizes, int n_in,
                              void* d_out, int out_size, void* d_ws, size_t ws_size,
                              hipStream_t stream) {
  const float* x     = (const float*)d_in[0];
  const float* wqkv  = (const float*)d_in[1];
  const float* wproj = (const float*)d_in[2];
  float* out = (float*)d_out;

  char* ws = (char*)d_ws;
  size_t off = 0;
  auto take = [&](size_t bytes) -> char* {
    char* p = ws + off;
    off += (bytes + 255) & ~(size_t)255;
    return p;
  };
  __bf16* xb  = (__bf16*)take((size_t)M_  * C_ * 2);
  __bf16* wqb = (__bf16*)take((size_t)O3_ * C_ * 2);
  __bf16* wpb = (__bf16*)take((size_t)C_  * C_ * 2);
  int8_t* qf  = (int8_t*)take((size_t)B_ * H_ * (N_/16) * 1024);
  int8_t* kf  = (int8_t*)take((size_t)B_ * H_ * (N_/16) * 1024);
  int8_t* vf  = (int8_t*)take((size_t)B_ * H_ * (N_/64) * 4 * 1024);
  __bf16* yb  = (__bf16*)take((size_t)M_ * C_ * 2);
  (void)ws_size; (void)in_sizes; (void)n_in; (void)out_size;

  cvt_f32_bf16_kernel<<<(M_ * C_ + 255) / 256, 256, 0, stream>>>(x, xb, M_ * C_);
  cvt_f32_bf16_kernel<<<(O3_ * C_ + 255) / 256, 256, 0, stream>>>(wqkv, wqb, O3_ * C_);
  cvt_f32_bf16_kernel<<<(C_ * C_ + 255) / 256, 256, 0, stream>>>(wproj, wpb, C_ * C_);

  qkv_wmma_kernel<<<dim3(M_ / 32, O3_ / 32), 32, 0, stream>>>(xb, wqb, qf, kf, vf);
  attn_kernel<<<B_ * H_ * (N_ / 128), 128, 0, stream>>>(qf, kf, vf, yb);
  proj_wmma_kernel<<<dim3(M_ / 32, C_ / 32), 32, 0, stream>>>(yb, wpb, out);
}
